// ENNModelWithSparsityControl_34943853920662
// MI455X (gfx1250) — matrix operand order repaired and checked by reading, not compiled, not run on hardware
//
#include <hip/hip_runtime.h>
#include <cmath>

// ---- problem constants (from reference) ----
#define NUM_NEURONS 131072
#define NUM_STATES  32
#define CDIM        16
#define NLAYERS     4
#define BATCHSZ     4

typedef __attribute__((ext_vector_type(16))) _Float16     v16h;
typedef __attribute__((ext_vector_type(8)))  float        v8f;
typedef __attribute__((ext_vector_type(4)))  unsigned int u32x4;
typedef __attribute__((ext_vector_type(8)))  int          i32x8;
typedef __attribute__((ext_vector_type(4)))  int          i32x4;

union V8F  { v8f  v; float    e[8];  };
union V16H { v16h v; _Float16 e[16]; };

// ============================================================
// TDM: issue a 16x32 f32 tile transfer between global and LDS.
// D# fields per CDNA5 ISA (2-D tensor,
// groups 2/3 zero-filled).  All inputs must be wave-uniform.
// ============================================================
__device__ __forceinline__ void tdm_tile_16x32_f32(
    unsigned long long gaddr, unsigned int lds_addr, bool is_store)
{
    u32x4 g0;
    g0[0] = 1u;                                   // count=1, user descriptor
    g0[1] = lds_addr;                             // LDS byte address
    g0[2] = (unsigned int)gaddr;                  // global_addr[31:0]
    g0[3] = ((unsigned int)(gaddr >> 32) & 0x01FFFFFFu) | (2u << 30); // [56:32] | type=2
    i32x8 g1;
    g1[0] = (int)(2u << 16);                      // data_size = 4B
    g1[1] = (int)(32u << 16);                     // tensor_dim0 = 32 (bits 79:48 lo)
    g1[2] = (int)(16u << 16);                     // tensor_dim1 = 16 (bits 111:80 lo)
    g1[3] = (int)(32u << 16);                     // tile_dim0 = 32 (bits 127:112)
    g1[4] = 16;                                   // tile_dim1 = 16, tile_dim2 = 0
    g1[5] = 32;                                   // tensor_dim0_stride = 32
    g1[6] = 0;
    g1[7] = 0;
    i32x4 z4 = {0, 0, 0, 0};
#if __clang_major__ >= 23
    i32x8 z8 = {0, 0, 0, 0, 0, 0, 0, 0};
    if (is_store) __builtin_amdgcn_tensor_store_from_lds(g0, g1, z4, z4, z8, 0);
    else          __builtin_amdgcn_tensor_load_to_lds  (g0, g1, z4, z4, z8, 0);
#else
    if (is_store) __builtin_amdgcn_tensor_store_from_lds(g0, g1, z4, z4, 0);
    else          __builtin_amdgcn_tensor_load_to_lds  (g0, g1, z4, z4, 0);
#endif
}

// ============================================================
// Kernel 1: x_out = tanh^4(x)   (pure streaming, HBM-bound)
// ============================================================
__global__ __launch_bounds__(256) void enn_tanh4_kernel(
    const float4* __restrict__ in, float4* __restrict__ out, int n4)
{
    int i      = blockIdx.x * blockDim.x + threadIdx.x;
    int stride = gridDim.x * blockDim.x;
    for (; i < n4; i += stride) {
        if (i + stride < n4) __builtin_prefetch(in + i + stride, 0, 0); // global_prefetch_b8
        float4 v = in[i];
        #pragma unroll
        for (int l = 0; l < NLAYERS; ++l) {
            v.x = tanhf(v.x); v.y = tanhf(v.y);
            v.z = tanhf(v.z); v.w = tanhf(v.w);
        }
        out[i] = v;
    }
}

// ============================================================
// Kernel 2: 4-layer state recurrence, 16 neurons per wave32.
// TDM loads the state tile into LDS; states then live in WMMA
// C-layout registers; C->A transposes via the wave-private LDS
// tile; GEMMs on v_wmma_f32_16x16x32_f16; TDM stores results.
// ============================================================
__global__ __launch_bounds__(256) void enn_states_kernel(
    const float* __restrict__ ns,   const float* __restrict__ Wenc,
    const float* __restrict__ benc, const float* __restrict__ Wdec,
    const float* __restrict__ bdec, float* __restrict__ out_states)
{
    // per-wave slice: 16x32 f32 tile (512) + 16 kth values
    __shared__ float smem[8 * 528];
    const int lane = threadIdx.x & 31;
    // readfirstlane -> provably wave-uniform so the TDM descriptor stays in SGPRs
    const int wave = __builtin_amdgcn_readfirstlane((int)(threadIdx.x >> 5));
    float* tile = smem + wave * 528;
    float* kthv = tile + 512;

    const int hi = lane >> 4;   // half-wave select
    const int lo = lane & 15;
    const int nbase = (blockIdx.x * 8 + wave) * 16; // first neuron of this wave

    // ---- kick off TDM DMA of this wave's 16x32 state tile into LDS ----
    const unsigned int tile_lds = (unsigned int)(uintptr_t)tile; // low 32b = LDS offset
    tdm_tile_16x32_f32(
        (unsigned long long)(uintptr_t)(ns + (size_t)nbase * NUM_STATES),
        tile_lds, /*is_store=*/false);

    // ---- weights into WMMA B-layout (K=32x16) while the TDM flies:
    //      lane lo = column N, half j holds B[K = j<8 ? 8*hi + j : 16 + 8*hi + (j-8)][N]; B[k][n] = W[n][k].
    V16H bE, bD0, bD1;
    #pragma unroll
    for (int j = 0; j < 16; ++j) {
        int k = (j < 8) ? (8 * hi + j) : (16 + 8 * hi + (j - 8));
        bE.e[j]  = (_Float16)Wenc[lo * NUM_STATES + k];
        bD0.e[j] = (k < CDIM) ? (_Float16)Wdec[(lo)      * CDIM + k] : (_Float16)0.0f;
        bD1.e[j] = (k < CDIM) ? (_Float16)Wdec[(16 + lo) * CDIM + k] : (_Float16)0.0f;
    }
    const float bencN = benc[lo];
    const float bdec0 = bdec[lo];
    const float bdec1 = bdec[16 + lo];

    // ---- TDM done -> pick up initial states in C layout:
    //      VGPR r, lane -> (M = r + 8*hi, N = lo)
    __builtin_amdgcn_s_wait_tensorcnt(0);
    V8F st0, st1, ac0, ac1;
    #pragma unroll
    for (int r = 0; r < 8; ++r) {
        int M = r + 8 * hi;
        st0.e[r] = tile[M * 32 + lo];
        st1.e[r] = tile[M * 32 + 16 + lo];
        ac0.e[r] = 0.f; ac1.e[r] = 0.f;
    }

    float wsum = 0.f;
    for (int layer = 0; layer < NLAYERS; ++layer) {
        // -- sparsity mask + decay
        #pragma unroll
        for (int r = 0; r < 8; ++r) {
            float a = st0.e[r]; st0.e[r] = (fabsf(a) >= 0.01f) ? a * 0.9f : 0.f;
            float b = st1.e[r]; st1.e[r] = (fabsf(b) >= 0.01f) ? b * 0.9f : 0.f;
        }
        // -- recency-weighted buffer collapses to running accumulator:
        //    acc_L = s_L + 0.7*acc_{L-1};  avg = acc_L / sum_{j<=L} 0.7^j
        wsum = 1.f + 0.7f * wsum;
        const float inv = 1.f / wsum;
        #pragma unroll
        for (int r = 0; r < 8; ++r) {
            ac0.e[r] = st0.e[r] + 0.7f * ac0.e[r];
            ac1.e[r] = st1.e[r] + 0.7f * ac1.e[r];
            st0.e[r] = ac0.e[r] * inv;
            st1.e[r] = ac1.e[r] * inv;
        }
        // -- stage averaged states (C-layout -> row-major f32 in LDS)
        #pragma unroll
        for (int r = 0; r < 8; ++r) {
            int M = r + 8 * hi;
            tile[M * 32 + lo]      = st0.e[r];
            tile[M * 32 + 16 + lo] = st1.e[r];
        }
        asm volatile("s_wait_dscnt 0" ::: "memory");
        // -- gather A-layout (f16 16x32): lane lo = row M; half j -> K
        V16H a;
        #pragma unroll
        for (int j = 0; j < 16; ++j) {
            int k = (j < 8) ? (8 * hi + j) : (16 + 8 * hi + (j - 8));
            a.e[j] = (_Float16)tile[lo * 32 + k];
        }
        v8f c = {};
        // h = states_avg @ Wenc^T   (single 16x16x32 WMMA)
        v8f hacc = __builtin_amdgcn_wmma_f32_16x16x32_f16(
            false, a.v, false, bE.v, (short)0, c, false, false);
        V8F hh; hh.v = hacc;
        #pragma unroll
        for (int r = 0; r < 8; ++r) hh.e[r] = fmaxf(hh.e[r] + bencN, 0.f); // +benc, ReLU

        // -- stage h (16x16) and rebuild A-layout with K padded to 32
        asm volatile("s_wait_dscnt 0" ::: "memory");
        #pragma unroll
        for (int r = 0; r < 8; ++r) {
            int M = r + 8 * hi;
            tile[M * 16 + lo] = hh.e[r];
        }
        asm volatile("s_wait_dscnt 0" ::: "memory");
        V16H a2;
        #pragma unroll
        for (int j = 0; j < 16; ++j) {
            if (j < 8) {
                int k = 8 * hi + j;           // k in [0,16): real data
                a2.e[j] = (_Float16)tile[lo * 16 + k];
            } else {
                a2.e[j] = (_Float16)0.0f;     // K = 16..31 zero padding
            }
        }
        // rec = h @ Wdec^T : two 16x16 output tiles (state cols 0-15, 16-31)
        v8f r0 = __builtin_amdgcn_wmma_f32_16x16x32_f16(
            false, a2.v, false, bD0.v, (short)0, c, false, false);
        v8f r1 = __builtin_amdgcn_wmma_f32_16x16x32_f16(
            false, a2.v, false, bD1.v, (short)0, c, false, false);
        V8F R0, R1; R0.v = r0; R1.v = r1;
        // -- +bdec, importance threshold
        #pragma unroll
        for (int r = 0; r < 8; ++r) {
            float v0 = R0.e[r] + bdec0;
            float v1 = R1.e[r] + bdec1;
            st0.e[r] = (fabsf(v0) >= 0.05f) ? v0 : 0.f;
            st1.e[r] = (fabsf(v1) >= 0.05f) ? v1 : 0.f;
        }
        // -- top-8 magnitude keep, per neuron row: stage rows to LDS
        #pragma unroll
        for (int r = 0; r < 8; ++r) {
            int M = r + 8 * hi;
            tile[M * 32 + lo]      = st0.e[r];
            tile[M * 32 + 16 + lo] = st1.e[r];
        }
        asm volatile("s_wait_dscnt 0" ::: "memory");
        {
            // all 32 lanes compute (lanes 16-31 duplicate rows 0-15: same
            // value written twice -> benign, keeps EXEC full for WMMA)
            float top[8];
            #pragma unroll
            for (int j = 0; j < 8; ++j) top[j] = -1.f;
            #pragma unroll 4
            for (int s = 0; s < 32; ++s) {
                float v = fabsf(tile[lo * 32 + s]);
                #pragma unroll
                for (int j = 0; j < 8; ++j) {   // sorted-insert network, regs only
                    float mx = fmaxf(top[j], v);
                    v        = fminf(top[j], v);
                    top[j]   = mx;
                }
            }
            kthv[lo] = top[7];  // 8th-largest |state|
        }
        asm volatile("s_wait_dscnt 0" ::: "memory");
        #pragma unroll
        for (int r = 0; r < 8; ++r) {
            int M = r + 8 * hi;
            float kth = kthv[M];
            st0.e[r] = (fabsf(st0.e[r]) >= kth) ? st0.e[r] : 0.f;
            st1.e[r] = (fabsf(st1.e[r]) >= kth) ? st1.e[r] : 0.f;
        }
    }
    // ---- stage final states to LDS and TDM-store the contiguous 2KB tile
    #pragma unroll
    for (int r = 0; r < 8; ++r) {
        int M = r + 8 * hi;
        tile[M * 32 + lo]      = st0.e[r];
        tile[M * 32 + 16 + lo] = st1.e[r];
    }
    asm volatile("s_wait_dscnt 0" ::: "memory");  // LDS data visible before TDM reads it
    tdm_tile_16x32_f32(
        (unsigned long long)(uintptr_t)(out_states + (size_t)nbase * NUM_STATES),
        tile_lds, /*is_store=*/true);
    __builtin_amdgcn_s_wait_tensorcnt(0);         // drain before wave exit
}

// ============================================================
extern "C" void kernel_launch(void* const* d_in, const int* in_sizes, int n_in,
                              void* d_out, int out_size, void* d_ws, size_t ws_size,
                              hipStream_t stream) {
    const float* x    = (const float*)d_in[0];
    const float* ns   = (const float*)d_in[1];
    const float* Wenc = (const float*)d_in[2];
    const float* benc = (const float*)d_in[3];
    const float* Wdec = (const float*)d_in[4];
    const float* bdec = (const float*)d_in[5];

    float* out_x      = (float*)d_out;
    float* out_states = out_x + (size_t)BATCHSZ * NUM_NEURONS * NUM_STATES;

    const int n4 = (BATCHSZ * NUM_NEURONS * NUM_STATES) / 4;
    enn_tanh4_kernel<<<4096, 256, 0, stream>>>((const float4*)x, (float4*)out_x, n4);

    // 16 neurons per wave, 8 waves per block -> 128 neurons/block
    enn_states_kernel<<<NUM_NEURONS / 128, 256, 0, stream>>>(
        ns, Wenc, benc, Wdec, bdec, out_states);
}
